// PackDynamicSequenceChunker_50903952392493
// MI455X (gfx1250) — matrix operand order repaired and checked by reading, not compiled, not run on hardware
//
#include <hip/hip_runtime.h>
#include <stdint.h>

#define B_SZ 4
#define L_SZ 4096
#define D_SZ 2048
#define M_SZ (B_SZ * L_SZ)   // 16384 rows
#define NQK  (2 * D_SZ)      // 4096
#define NSEG 4
#define SEGL (L_SZ / NSEG)   // 1024

typedef __attribute__((ext_vector_type(16))) __bf16 bf16x16;
typedef __attribute__((ext_vector_type(8)))  __bf16 bf16x8;
typedef __attribute__((ext_vector_type(8)))  float  f32x8;

// ---------------------------------------------------------------------------
// fp32 -> (bf16 hi, bf16 lo) split, round-to-nearest-even
// ---------------------------------------------------------------------------
__device__ __forceinline__ void bf16_split(float x, unsigned short &hi, unsigned short &lo) {
  union { float f; unsigned u; } c; c.f = x;
  unsigned r = c.u + 0x7FFFu + ((c.u >> 16) & 1u);
  hi = (unsigned short)(r >> 16);
  union { unsigned u; float f; } h; h.u = ((unsigned)hi) << 16;
  float rem = x - h.f;
  union { float f; unsigned u; } cl; cl.f = rem;
  unsigned rl = cl.u + 0x7FFFu + ((cl.u >> 16) & 1u);
  lo = (unsigned short)(rl >> 16);
}

// ---------------------------------------------------------------------------
// bf16x3-split WMMA GEMM: C = A(MxK) * B(KxN) (+ bias), fp32 in/out.
// 128x128 block tile, K-step 32, 8 waves; each wave = 32x64 (2x4 WMMA tiles).
// ---------------------------------------------------------------------------
#define TM 128
#define TN 128
#define TK 32
#define LDT 48   // padded LDS stride (elements); 96B rows keep 16B alignment

__global__ __launch_bounds__(256)
void gemm_bf16x3_wmma(const float* __restrict__ A, const float* __restrict__ Bw,
                      const float* __restrict__ bias, float* __restrict__ C,
                      int M, int N, int K) {
  __shared__ alignas(16) unsigned short sAhi[TM][LDT];
  __shared__ alignas(16) unsigned short sAlo[TM][LDT];
  __shared__ alignas(16) unsigned short sBhi[TN][LDT];
  __shared__ alignas(16) unsigned short sBlo[TN][LDT];

  const int tid  = threadIdx.x;
  const int wid  = tid >> 5;
  const int lane = tid & 31;
  const int wm = wid & 3;       // M wave index: 0..3, 32 rows each
  const int wn = wid >> 2;      // N wave index: 0..1, 64 cols each
  const int kh = lane >> 4;     // half-wave (selects K half per ISA layout)
  const int lm = lane & 15;

  const int n0 = blockIdx.x * TN;
  const int m0 = blockIdx.y * TM;

  const f32x8 vzero = {0.f,0.f,0.f,0.f,0.f,0.f,0.f,0.f};
  f32x8 acc[2][4];
  #pragma unroll
  for (int i = 0; i < 2; ++i)
    #pragma unroll
    for (int j = 0; j < 4; ++j) acc[i][j] = vzero;

  // staging thread mapping
  const int arow  = tid >> 1;   // 0..127
  const int ahalf = tid & 1;    // two 16-float halves of the 32-wide K slab
  const int bkrow = tid >> 3;   // 0..31
  const int bseg  = tid & 7;    // eight 16-col segments

  for (int kb = 0; kb < K; kb += TK) {
    if (kb + TK < K) {  // prefetch next slabs (global_prefetch_b8)
      __builtin_prefetch(A + (size_t)(m0 + arow) * K + kb + TK + ahalf * 16, 0, 0);
      __builtin_prefetch(Bw + (size_t)(kb + TK + bkrow) * N + n0 + bseg * 16, 0, 0);
    }
    // ---- stage A tile (TM x TK), split fp32 -> bf16 hi/lo ----
    {
      const float4* src = (const float4*)(A + (size_t)(m0 + arow) * K + kb + ahalf * 16);
      #pragma unroll
      for (int v = 0; v < 4; ++v) {
        float4 f = src[v];
        float e[4] = {f.x, f.y, f.z, f.w};
        #pragma unroll
        for (int j = 0; j < 4; ++j) {
          unsigned short h, l; bf16_split(e[j], h, l);
          int kk = ahalf * 16 + v * 4 + j;
          sAhi[arow][kk] = h; sAlo[arow][kk] = l;
        }
      }
    }
    // ---- stage B tile (TK x TN) into column-major [n][k] ----
    {
      const float4* src = (const float4*)(Bw + (size_t)(kb + bkrow) * N + n0 + bseg * 16);
      #pragma unroll
      for (int v = 0; v < 4; ++v) {
        float4 f = src[v];
        float e[4] = {f.x, f.y, f.z, f.w};
        #pragma unroll
        for (int j = 0; j < 4; ++j) {
          unsigned short h, l; bf16_split(e[j], h, l);
          int nn = bseg * 16 + v * 4 + j;
          sBhi[nn][bkrow] = h; sBlo[nn][bkrow] = l;
        }
      }
    }
    __syncthreads();

    // ---- load fragments per ISA 16-bit A/B VGPR layouts ----
    bf16x16 fahi[2], falo[2], fbhi[4], fblo[4];
    #pragma unroll
    for (int mi = 0; mi < 2; ++mi) {
      int row = wm * 32 + mi * 16 + lm;
      int c0 = kh * 8, c1 = 16 + kh * 8;
      bf16x8 h0 = *(const bf16x8*)&sAhi[row][c0];
      bf16x8 h1 = *(const bf16x8*)&sAhi[row][c1];
      fahi[mi] = __builtin_shufflevector(h0, h1, 0,1,2,3,4,5,6,7,8,9,10,11,12,13,14,15);
      bf16x8 l0 = *(const bf16x8*)&sAlo[row][c0];
      bf16x8 l1 = *(const bf16x8*)&sAlo[row][c1];
      falo[mi] = __builtin_shufflevector(l0, l1, 0,1,2,3,4,5,6,7,8,9,10,11,12,13,14,15);
    }
    #pragma unroll
    for (int ni = 0; ni < 4; ++ni) {
      int col = wn * 64 + ni * 16 + lm;
      int kb2 = kh * 8;
      bf16x8 h0 = *(const bf16x8*)&sBhi[col][kb2];
      bf16x8 h1 = *(const bf16x8*)&sBhi[col][kb2 + 16];
      fbhi[ni] = __builtin_shufflevector(h0, h1, 0,1,2,3,4,5,6,7,8,9,10,11,12,13,14,15);
      bf16x8 l0 = *(const bf16x8*)&sBlo[col][kb2];
      bf16x8 l1 = *(const bf16x8*)&sBlo[col][kb2 + 16];
      fblo[ni] = __builtin_shufflevector(l0, l1, 0,1,2,3,4,5,6,7,8,9,10,11,12,13,14,15);
    }

    // ---- 3-term split product: hi*hi + hi*lo + lo*hi ----
    #pragma unroll
    for (int mi = 0; mi < 2; ++mi) {
      #pragma unroll
      for (int ni = 0; ni < 4; ++ni) {
        acc[mi][ni] = __builtin_amdgcn_wmma_f32_16x16x32_bf16(
            false, fahi[mi], false, fbhi[ni], (short)0, acc[mi][ni], false, false);
        acc[mi][ni] = __builtin_amdgcn_wmma_f32_16x16x32_bf16(
            false, fahi[mi], false, fblo[ni], (short)0, acc[mi][ni], false, false);
        acc[mi][ni] = __builtin_amdgcn_wmma_f32_16x16x32_bf16(
            false, falo[mi], false, fbhi[ni], (short)0, acc[mi][ni], false, false);
      }
    }
    __syncthreads();
  }

  // ---- epilogue: C/D layout — VGPR r: lanes0-15 M=r, lanes16-31 M=8+r ----
  #pragma unroll
  for (int mi = 0; mi < 2; ++mi) {
    #pragma unroll
    for (int ni = 0; ni < 4; ++ni) {
      int gc  = n0 + wn * 64 + ni * 16 + lm;
      int gr0 = m0 + wm * 32 + mi * 16 + kh * 8;
      float bv = bias ? bias[gc] : 0.0f;
      #pragma unroll
      for (int r = 0; r < 8; ++r)
        C[(size_t)(gr0 + r) * N + gc] = acc[mi][ni][r] + bv;
    }
  }
}

// ---------------------------------------------------------------------------
// probs[b,t] = 0.5*(1 - cos(q_t, k_{t-1})); one wave per row
// ---------------------------------------------------------------------------
__global__ __launch_bounds__(256)
void probs_kernel(const float* __restrict__ qk, const float* __restrict__ start_key,
                  float* __restrict__ probs) {
  int wid = threadIdx.x >> 5, lane = threadIdx.x & 31;
  int row = blockIdx.x * 8 + wid;
  int t = row & (L_SZ - 1);
  const float* q  = qk + (size_t)row * NQK;
  const float* kp = (t == 0) ? start_key : (qk + (size_t)(row - 1) * NQK + D_SZ);
  float num = 0.f, qq = 0.f, kk = 0.f;
  for (int i = lane; i < D_SZ; i += 32) {
    float qi = q[i], ki = kp[i];
    num += qi * ki; qq += qi * qi; kk += ki * ki;
  }
  #pragma unroll
  for (int off = 16; off; off >>= 1) {
    num += __shfl_xor(num, off, 32);
    qq  += __shfl_xor(qq,  off, 32);
    kk  += __shfl_xor(kk,  off, 32);
  }
  if (lane == 0) {
    float den = fmaxf(sqrtf(qq) * sqrtf(kk), 1e-8f);
    probs[row] = (1.0f - num / den) * 0.5f;
  }
}

// ---------------------------------------------------------------------------
// Per-batch: boundary, cumsum (LDS scan), packed-metadata scatter,
// chunk_lens, bprobs, segmented gate-prefix products P, aux loss term.
// ---------------------------------------------------------------------------
__global__ __launch_bounds__(1024)
void boundary_kernel(const float* __restrict__ probs, float* __restrict__ bprobs,
                     float* __restrict__ Pprod, int* __restrict__ chunkid,
                     int* __restrict__ dest, int* __restrict__ lensOut,
                     float* __restrict__ auxb) {
  __shared__ float sBp[L_SZ];
  __shared__ int   sStart[L_SZ];
  __shared__ int   sScan[1024];
  __shared__ float sRed[1024];
  const int b = blockIdx.x, tid = threadIdx.x, t0 = tid * 4;

  float p[4]; int bnd[4];
  #pragma unroll
  for (int j = 0; j < 4; ++j) {
    p[j] = probs[b * L_SZ + t0 + j];
    bnd[j] = (p[j] > 0.5f) || (t0 + j == 0);
    sBp[t0 + j] = 0.0f;
  }
  int ls = bnd[0] + bnd[1] + bnd[2] + bnd[3];
  sScan[tid] = ls;
  __syncthreads();
  for (int off = 1; off < 1024; off <<= 1) {
    int v = sScan[tid];
    int add = (tid >= off) ? sScan[tid - off] : 0;
    __syncthreads();
    sScan[tid] = v + add;
    __syncthreads();
  }
  const int nch = sScan[1023];
  int cum = sScan[tid] - ls;   // exclusive prefix for this thread
  int cid[4];
  #pragma unroll
  for (int j = 0; j < 4; ++j) {
    cum += bnd[j];
    cid[j] = cum - 1;
    chunkid[b * L_SZ + t0 + j] = cum - 1;
    dest[b * L_SZ + t0 + j] = bnd[j] ? (cum - 1) : -1;
  }
  __syncthreads();
  #pragma unroll
  for (int j = 0; j < 4; ++j)
    if (bnd[j]) { sBp[cid[j]] = p[j]; sStart[cid[j]] = t0 + j; }
  __syncthreads();

  #pragma unroll
  for (int j = 0; j < 4; ++j) {
    int c = t0 + j, len = 0;
    if (c < nch) {
      int nxt = (c + 1 < nch) ? sStart[c + 1] : L_SZ;
      len = nxt - sStart[c];
    }
    lensOut[b * L_SZ + c] = len;
    bprobs[b * L_SZ + c]  = sBp[c];
  }

  // segmented inclusive product of gates g = 1 - bprobs (256 threads/segment)
  float g[4], lp = 1.0f;
  #pragma unroll
  for (int j = 0; j < 4; ++j) { g[j] = 1.0f - sBp[t0 + j]; lp *= g[j]; }
  int ltid = tid & 255;
  sRed[tid] = lp;
  __syncthreads();
  for (int off = 1; off < 256; off <<= 1) {
    float v = sRed[tid];
    float m = (ltid >= off) ? sRed[tid - off] : 1.0f;
    __syncthreads();
    sRed[tid] = v * m;
    __syncthreads();
  }
  float run = (ltid > 0) ? sRed[tid - 1] : 1.0f;
  #pragma unroll
  for (int j = 0; j < 4; ++j) { run *= g[j]; Pprod[b * L_SZ + t0 + j] = run; }
  __syncthreads();

  // aux loss term: F = nch/L (straight-through hard), G = mean(probs)
  sRed[tid] = p[0] + p[1] + p[2] + p[3];
  __syncthreads();
  for (int off = 512; off; off >>= 1) {
    if (tid < off) sRed[tid] += sRed[tid + off];
    __syncthreads();
  }
  if (tid == 0) {
    float G = sRed[0] / (float)L_SZ;
    float F = (float)nch / (float)L_SZ;
    const float Nt = 6.0f;
    auxb[b] = Nt / (Nt - 1.0f) * ((Nt - 1.0f) * F * G + (1.0f - F) * (1.0f - G));
  }
}

// ---------------------------------------------------------------------------
// Scatter-pack: downsampled[b, dest, :] = tokens[b,t,:] * probs[b,t]
// ---------------------------------------------------------------------------
__global__ __launch_bounds__(256)
void pack_kernel(const float* __restrict__ tokens, const float* __restrict__ probs,
                 const int* __restrict__ dest, float* __restrict__ down) {
  int t = blockIdx.x, b = blockIdx.y;
  int d0 = dest[b * L_SZ + t];
  if (d0 < 0) return;
  float pv = probs[b * L_SZ + t];
  const float4* src = (const float4*)(tokens + (size_t)(b * L_SZ + t) * D_SZ);
  float4* dst = (float4*)(down + (size_t)(b * L_SZ + d0) * D_SZ);
  for (int i = threadIdx.x; i < D_SZ / 4; i += 256) {
    float4 v = src[i];
    v.x *= pv; v.y *= pv; v.z *= pv; v.w *= pv;
    dst[i] = v;
  }
}

// ---------------------------------------------------------------------------
// Segment-local gated scan: h = h*g_t + x_t (one thread per channel)
// ---------------------------------------------------------------------------
__global__ __launch_bounds__(256)
void scan_local_kernel(const float* __restrict__ down, const float* __restrict__ bprobs,
                       float* __restrict__ scanned, float* __restrict__ lastv) {
  __shared__ float sG[256];
  int d = blockIdx.x * 256 + threadIdx.x;
  int s = blockIdx.y, b = blockIdx.z;
  float h = 0.0f;
  for (int c = 0; c < SEGL; c += 256) {
    __syncthreads();
    sG[threadIdx.x] = 1.0f - bprobs[b * L_SZ + s * SEGL + c + threadIdx.x];
    __syncthreads();
    for (int u = 0; u < 256; ++u) {
      size_t idx = ((size_t)(b * L_SZ + s * SEGL + c + u)) * D_SZ + d;
      h = h * sG[u] + down[idx];
      scanned[idx] = h;
    }
  }
  lastv[(b * NSEG + s) * D_SZ + d] = h;
}

// carry across segments: H_s = last_{s-1} + Pend_{s-1} * H_{s-1}
__global__ __launch_bounds__(256)
void carry_kernel(const float* __restrict__ lastv, const float* __restrict__ Pprod,
                  float* __restrict__ H) {
  int idx = blockIdx.x * 256 + threadIdx.x;
  int b = idx / D_SZ, d = idx % D_SZ;
  float hc = 0.0f;
  H[(b * NSEG + 0) * D_SZ + d] = 0.0f;
  #pragma unroll
  for (int s = 1; s < NSEG; ++s) {
    hc = lastv[(b * NSEG + s - 1) * D_SZ + d] + Pprod[b * L_SZ + s * SEGL - 1] * hc;
    H[(b * NSEG + s) * D_SZ + d] = hc;
  }
}

// ---------------------------------------------------------------------------
// upsampled[b,t,:] += scanned_local[b,cid,:] + P[b,cid] * H[b,seg(cid),:]
// (residual GEMM result + bias already resident in `up`)
// ---------------------------------------------------------------------------
__global__ __launch_bounds__(256)
void upsample_kernel(const float* __restrict__ scanned, const float* __restrict__ Pprod,
                     const float* __restrict__ H, const int* __restrict__ chunkid,
                     float* __restrict__ up) {
  int t = blockIdx.x, b = blockIdx.y;
  int cid = chunkid[b * L_SZ + t];
  int sg = cid / SEGL;
  float pc = Pprod[b * L_SZ + cid];
  const float4* sc = (const float4*)(scanned + (size_t)(b * L_SZ + cid) * D_SZ);
  const float4* Hv = (const float4*)(H + (b * NSEG + sg) * D_SZ);
  float4* u = (float4*)(up + (size_t)(b * L_SZ + t) * D_SZ);
  for (int i = threadIdx.x; i < D_SZ / 4; i += 256) {
    float4 s4 = sc[i], h4 = Hv[i], u4 = u[i];
    u4.x += s4.x + pc * h4.x;
    u4.y += s4.y + pc * h4.y;
    u4.z += s4.z + pc * h4.z;
    u4.w += s4.w + pc * h4.w;
    u[i] = u4;
  }
}

__global__ void aux_kernel(const float* __restrict__ auxb, float* __restrict__ out) {
  if (threadIdx.x == 0)
    out[0] = (auxb[0] + auxb[1] + auxb[2] + auxb[3]) * 0.25f * 0.03f;
}

// ---------------------------------------------------------------------------
extern "C" void kernel_launch(void* const* d_in, const int* in_sizes, int n_in,
                              void* d_out, int out_size, void* d_ws, size_t ws_size,
                              hipStream_t stream) {
  (void)in_sizes; (void)n_in; (void)out_size; (void)ws_size;
  const float* tokens    = (const float*)d_in[0];
  const float* W_qk      = (const float*)d_in[1];
  const float* start_key = (const float*)d_in[2];
  const float* W_res     = (const float*)d_in[3];
  const float* b_res     = (const float*)d_in[4];

  float* outDown = (float*)d_out;
  float* outUp   = outDown + (size_t)M_SZ * D_SZ;
  float* outAux  = outUp   + (size_t)M_SZ * D_SZ;
  int*   outLens = (int*)(outAux + 1);

  char* ws = (char*)d_ws;
  size_t off = 0;
  float* qk = (float*)(ws + off);                // M x 4096
  float* scanned = qk;                           // reuse qk region after probs
  off += (size_t)M_SZ * NQK * sizeof(float);
  float* probs  = (float*)(ws + off); off += (size_t)M_SZ * sizeof(float);
  float* bprobs = (float*)(ws + off); off += (size_t)M_SZ * sizeof(float);
  float* Pprod  = (float*)(ws + off); off += (size_t)M_SZ * sizeof(float);
  int*   chunkid= (int*)  (ws + off); off += (size_t)M_SZ * sizeof(int);
  int*   dest   = (int*)  (ws + off); off += (size_t)M_SZ * sizeof(int);
  float* lastv  = (float*)(ws + off); off += (size_t)B_SZ * NSEG * D_SZ * sizeof(float);
  float* Hbuf   = (float*)(ws + off); off += (size_t)B_SZ * NSEG * D_SZ * sizeof(float);
  float* auxb   = (float*)(ws + off); off += 4 * sizeof(float);

  // zero the downsampled output region (scatter target)
  hipMemsetAsync(outDown, 0, (size_t)M_SZ * D_SZ * sizeof(float), stream);

  dim3 blk(256);
  // GEMM 1: qk = tokens @ W_qk           (16384 x 2048 x 4096)
  gemm_bf16x3_wmma<<<dim3(NQK / TN, M_SZ / TM), blk, 0, stream>>>(
      tokens, W_qk, nullptr, qk, M_SZ, NQK, D_SZ);
  // GEMM 2: outUp = tokens @ W_res + b_res  (written straight into upsampled)
  gemm_bf16x3_wmma<<<dim3(D_SZ / TN, M_SZ / TM), blk, 0, stream>>>(
      tokens, W_res, b_res, outUp, M_SZ, D_SZ, D_SZ);

  probs_kernel<<<M_SZ / 8, 256, 0, stream>>>(qk, start_key, probs);
  boundary_kernel<<<B_SZ, 1024, 0, stream>>>(probs, bprobs, Pprod, chunkid, dest,
                                             outLens, auxb);
  pack_kernel<<<dim3(L_SZ, B_SZ), 256, 0, stream>>>(tokens, probs, dest, outDown);
  scan_local_kernel<<<dim3(D_SZ / 256, NSEG, B_SZ), 256, 0, stream>>>(
      outDown, bprobs, scanned, lastv);
  carry_kernel<<<(B_SZ * D_SZ) / 256, 256, 0, stream>>>(lastv, Pprod, Hbuf);
  upsample_kernel<<<dim3(L_SZ, B_SZ), 256, 0, stream>>>(scanned, Pprod, Hbuf,
                                                        chunkid, outUp);
  aux_kernel<<<1, 32, 0, stream>>>(auxb, outAux);
}
